// SingleHeadPhraseCompressor_68015102099509
// MI455X (gfx1250) — matrix operand order repaired and checked by reading, not compiled, not run on hardware
//
#include <hip/hip_runtime.h>
#include <hip/hip_bf16.h>

typedef _Float16 v16h __attribute__((ext_vector_type(16)));
typedef float    v8f  __attribute__((ext_vector_type(8)));

#define BB 4
#define TT 4096
#define DD 2048
#define PP 1024
#define LL 8
#define HH 64
#define NN 128            // 2*H  (W_kv columns | W_z columns)
#define KSTEPS 64         // DD / 32 WMMA k-steps total
#define MT 4              // M-tiles (16 rows each) per workgroup
#define ROWS 64           // MT * 16 gathered rows per workgroup
#define PHW 8             // phrases per workgroup
#define KHALF_STEPS 32    // k-steps per K-half (wave pair split)
#define CH_KSTEPS 4       // k-steps per chunk per K-half
#define KCHUNK_HALF 128   // k values per chunk per K-half
#define NCHUNKS 8

// ---------------------------------------------------------------------------
// Kernel 1: pre-convert [W_kv ; W_z] (f32, 128 x 2048) into f16 laid out in
// exact WMMA-B fragment order: Wf[nt][ks][lane][16 halves].
// B layout (16-bit 32x16): lane<16 -> col n=lane&15, K = 0..7,16..23 ; lane>=16 -> K = 8..15,24..31.
// ---------------------------------------------------------------------------
__global__ __launch_bounds__(256) void prep_w_kernel(
    const float* __restrict__ Wkv, const float* __restrict__ Wz,
    _Float16* __restrict__ Wf)
{
  int q    = blockIdx.x * 256 + threadIdx.x;   // 65536 quads of 4 halves
  int qi   = q & 3;
  int lane = (q >> 2) & 31;
  int ks   = (q >> 7) & 63;
  int nt   = q >> 13;                          // 0..7  (N-tile)
  int nrow = nt * 16 + (lane & 15);            // global N column = W row
  int hb   = qi * 4;                           // half base within lane (0,4,8,12)
  int k    = ks * 32 + (lane >> 4) * 8 + (hb & 4) + (hb >> 3) * 16;  // table-order K per half-wave
  const float* src = (nrow < HH) ? (Wkv + (size_t)nrow * DD + k)
                                 : (Wz  + (size_t)(nrow - HH) * DD + k);
  float4 v = *(const float4*)src;
  union { _Float16 h[4]; uint2 u; } cv;
  cv.h[0] = (_Float16)v.x; cv.h[1] = (_Float16)v.y;
  cv.h[2] = (_Float16)v.z; cv.h[3] = (_Float16)v.w;
  *(uint2*)(Wf + ((size_t)((nt * 64 + ks) * 32 + lane)) * 16 + hb) = cv.u;
}

// ---------------------------------------------------------------------------
// Kernel 2: one workgroup (8 waves, 256 threads) per 8 phrases (M = 64).
//   Wave w: khalf = w>>2 (K-range split), owns N-tiles 2*(w&3), 2*(w&3)+1.
//   Per k-step: 4 A frags + 2 B frags -> 8 independent WMMAs
//   (1 ds_load_b128 + 0.5 global_load_b128 per v_wmma).
//   K-halves reduced through the LDS C tile at the end.
// ---------------------------------------------------------------------------
__global__ __launch_bounds__(256) void phrase_compress_kernel(
    const float*          __restrict__ hbuf,   // (B,T,D) f32
    const unsigned char*  __restrict__ pmask,  // (B,P,L) bool
    const long long*      __restrict__ pidx,   // (B,P,L) int64
    const float*          __restrict__ Bpos,   // (L,H) f32
    const _Float16*       __restrict__ Wf,     // fragment-ordered weights
    float*                __restrict__ out)    // (B,P,H) f32
{
  // 32 KB buffer: A fragments during the K loop, then the 64x128 f32 C tile.
  __shared__ __align__(32) unsigned char shBuf[32 * 1024];
  __shared__ int shTok[ROWS];
  __shared__ unsigned char shMask[ROWS];
  _Float16* shA = (_Float16*)shBuf;   // [khalf][mt][ksl][lane][16 halves]
  float*    shC = (float*)shBuf;      // [row(64)][col(128)]

  const int tid = threadIdx.x;
  const int fp0 = blockIdx.x * PHW;   // flat phrase base (b*P + p)

  if (tid < ROWS) {
    int fp = fp0 + (tid >> 3);
    int l  = tid & 7;
    shTok[tid]  = (int)pidx[(size_t)fp * LL + l];
    shMask[tid] = pmask[(size_t)fp * LL + l];
  }
  __syncthreads();

  // ---- chunk-invariant staging constants: thread owns row r, quad-lane sub ----
  // A layout (16-bit 16x32): lane = row + 16*laneHi; laneHi=(kk>>3)&1;
  // half = (kk&7) + 8*(kk>>4). With kl = sub*4 + j*16 (j=0..7 per K-half):
  //   laneHi = sub>>1, half = (sub&1)*4 + 8*(j&1), ksl = j>>1.
  const int r     = tid >> 2;          // gathered row 0..63
  const int sub   = tid & 3;
  const int mt_st = r >> 4;
  const int rt    = r & 15;
  const float* srcRow;
  {
    int fp = fp0 + (r >> 3);
    int b  = fp >> 10;                 // fp / P
    srcRow = hbuf + ((size_t)b * TT + shTok[r]) * DD;
  }
  const int lane_st = rt + ((sub >> 1) << 4);
  // half-index base into shA: [khalf]*8192 + mt*2048 + ksl*512 + lane*16 + half
  const int ldsBase = mt_st * 2048 + lane_st * 16 + (sub & 1) * 4;

  const int wave  = tid >> 5;
  const int ln    = tid & 31;
  const int khalf = wave >> 2;         // which K half this wave computes
  const int ntA   = (wave & 3) * 2;    // first of two owned N-tiles

  v8f acc[2][MT];
  #pragma unroll
  for (int n2 = 0; n2 < 2; ++n2)
    #pragma unroll
    for (int mt = 0; mt < MT; ++mt) acc[n2][mt] = (v8f){};

  const v16h* __restrict__ Af = ((const v16h*)shA) + khalf * 512 + ln;
  const v16h* __restrict__ B0 =
      ((const v16h*)Wf) + ((size_t)ntA * KSTEPS + (size_t)khalf * KHALF_STEPS) * 32 + ln;
  const v16h* __restrict__ B1 = B0 + (size_t)KSTEPS * 32;  // next N-tile

  for (int chunk = 0; chunk < NCHUNKS; ++chunk) {
    const int k0 = chunk * KCHUNK_HALF;
    // ---- stage this thread's row: 8 quads for each K-half window ----
    #pragma unroll
    for (int w2 = 0; w2 < 2; ++w2) {
      #pragma unroll
      for (int j = 0; j < 8; ++j) {
        float4 v = *(const float4*)(srcRow + w2 * 1024 + k0 + sub * 4 + j * 16);
        union { _Float16 h[4]; uint2 u; } cv;
        cv.h[0] = (_Float16)v.x; cv.h[1] = (_Float16)v.y;
        cv.h[2] = (_Float16)v.z; cv.h[3] = (_Float16)v.w;
        *(uint2*)(shA + ldsBase + w2 * 8192 + (j >> 1) * 512 + (j & 1) * 8) = cv.u;
      }
    }
    __syncthreads();

    // ---- compute: 4 k-steps; 4 A frags + 2 B frags -> 8 WMMAs each ----
    const v16h* __restrict__ B0c = B0 + (size_t)(chunk * CH_KSTEPS) * 32;
    const v16h* __restrict__ B1c = B1 + (size_t)(chunk * CH_KSTEPS) * 32;
    #pragma unroll 2
    for (int ksl = 0; ksl < CH_KSTEPS; ++ksl) {
      v16h b0 = B0c[ksl * 32];
      v16h b1 = B1c[ksl * 32];
      v16h a0 = Af[(0 * 4 + ksl) * 32];   // v16h index: mt*128 + ksl*32
      v16h a1 = Af[(1 * 4 + ksl) * 32];
      v16h a2 = Af[(2 * 4 + ksl) * 32];
      v16h a3 = Af[(3 * 4 + ksl) * 32];
      acc[0][0] = __builtin_amdgcn_wmma_f32_16x16x32_f16(false, a0, false, b0, (short)0, acc[0][0], false, false);
      acc[0][1] = __builtin_amdgcn_wmma_f32_16x16x32_f16(false, a1, false, b0, (short)0, acc[0][1], false, false);
      acc[0][2] = __builtin_amdgcn_wmma_f32_16x16x32_f16(false, a2, false, b0, (short)0, acc[0][2], false, false);
      acc[0][3] = __builtin_amdgcn_wmma_f32_16x16x32_f16(false, a3, false, b0, (short)0, acc[0][3], false, false);
      acc[1][0] = __builtin_amdgcn_wmma_f32_16x16x32_f16(false, a0, false, b1, (short)0, acc[1][0], false, false);
      acc[1][1] = __builtin_amdgcn_wmma_f32_16x16x32_f16(false, a1, false, b1, (short)0, acc[1][1], false, false);
      acc[1][2] = __builtin_amdgcn_wmma_f32_16x16x32_f16(false, a2, false, b1, (short)0, acc[1][2], false, false);
      acc[1][3] = __builtin_amdgcn_wmma_f32_16x16x32_f16(false, a3, false, b1, (short)0, acc[1][3], false, false);
    }
    __syncthreads();
  }

  // ---- K-half reduction through the LDS C tile ----
  // C layout per 16x16 tile: VGPR j -> row j (lanes<16) / row j+8 (lanes>=16).
  const int colBase = ntA * 16 + (ln & 15);
  const int rowOff  = (ln >> 4) * 8;
  if (khalf == 1) {
    #pragma unroll
    for (int n2 = 0; n2 < 2; ++n2)
      #pragma unroll
      for (int mt = 0; mt < MT; ++mt)
        #pragma unroll
        for (int j = 0; j < 8; ++j)
          shC[(mt * 16 + rowOff + j) * NN + colBase + n2 * 16] = acc[n2][mt][j];
  }
  __syncthreads();
  if (khalf == 0) {
    #pragma unroll
    for (int n2 = 0; n2 < 2; ++n2)
      #pragma unroll
      for (int mt = 0; mt < MT; ++mt)
        #pragma unroll
        for (int j = 0; j < 8; ++j) {
          int idx = (mt * 16 + rowOff + j) * NN + colBase + n2 * 16;
          shC[idx] += acc[n2][mt][j];
        }
  }
  __syncthreads();

  // ---- epilogue: per-channel masked softmax over Lmax, gated sum ----
  // 8 phrases x 64 channels = 512 outputs, 2 per thread.
  #pragma unroll
  for (int it = 0; it < 2; ++it) {
    int idx = tid + it * 256;
    int ph  = idx >> 6;      // phrase within workgroup
    int hh  = idx & 63;      // channel
    int fp  = fp0 + ph;
    float zv[LL], cvv[LL];
    bool  mv[LL];
    float zmax = -3.0e38f;
    bool  any  = false;
    #pragma unroll
    for (int l = 0; l < LL; ++l) {
      int rr  = ph * 8 + l;
      mv[l]  = (shMask[rr] != 0);
      zv[l]  = shC[rr * NN + HH + hh] + Bpos[l * HH + hh];
      cvv[l] = shC[rr * NN + hh];
      if (mv[l]) { any = true; zmax = fmaxf(zmax, zv[l]); }
    }
    float s = 0.f, o = 0.f;
    #pragma unroll
    for (int l = 0; l < LL; ++l) {
      float e = mv[l] ? __expf(zv[l] - zmax) : 0.f;
      s += e; o += e * cvv[l];
    }
    out[(size_t)fp * HH + hh] = any ? (o / s) : 0.f;
  }
}

// ---------------------------------------------------------------------------
extern "C" void kernel_launch(void* const* d_in, const int* in_sizes, int n_in,
                              void* d_out, int out_size, void* d_ws, size_t ws_size,
                              hipStream_t stream)
{
  const float*         hbuf  = (const float*)d_in[0];         // (4,4096,2048) f32
  const unsigned char* pmask = (const unsigned char*)d_in[1]; // (4,1024,8) bool
  const long long*     pidx  = (const long long*)d_in[2];     // (4,1024,8) int64
  const float*         Wkv   = (const float*)d_in[3];         // (64,2048) f32
  const float*         Wz    = (const float*)d_in[4];         // (64,2048) f32
  const float*         Bpos  = (const float*)d_in[5];         // (8,64) f32
  float*               out   = (float*)d_out;                 // (4,1024,64) f32
  _Float16*            Wf    = (_Float16*)d_ws;               // 512 KB fragment weights

  prep_w_kernel<<<256, 256, 0, stream>>>(Wkv, Wz, Wf);
  phrase_compress_kernel<<<(BB * PP) / PHW, 256, 0, stream>>>(
      hbuf, pmask, pidx, Bpos, Wf, out);
}
